// MHDecoderTransformer_81905026335131
// MI455X (gfx1250) — matrix operand order repaired and checked by reading, not compiled, not run on hardware
//
#include <hip/hip_runtime.h>
#include <math.h>

// MI455X / gfx1250 implementation.
// All S*D-shaped GEMMs + both attention einsums run on v_wmma_f32_16x16x32_f16
// (f16 inputs, f32 accumulate). Softmax / layernorm / RBF / the 524MB-bound
// classifier GEMM stay fp32 (bandwidth-bound; precision-sensitive).

#define B_  32
#define S_  256
#define D_  256
#define H_  8
#define L_  4
#define V_  2000
#define C_  50
#define HD_ 32

typedef _Float16 half_t;
typedef __attribute__((ext_vector_type(16))) _Float16 v16h;
typedef __attribute__((ext_vector_type(8)))  _Float16 v8h;
typedef __attribute__((ext_vector_type(8)))  float    v8f;

union V16U { v16h v; v8h h[2]; };

// ---- WMMA fragment loaders (CDNA5 16-bit layouts, wave32) ------------------
// A 16x32: lanes 0-15 -> M=lane, K chunks {0..7, 16..23}; lanes 16-31 -> K chunks {8..15, 24..31}
__device__ inline v16h load_a16(const half_t* A, int lda, int mbase, int kbase, int lane) {
  int m  = mbase + (lane & 15);
  int k0 = kbase + ((lane >> 4) << 3);
  V16U u;
  u.h[0] = *(const v8h*)(A + m * lda + k0);
  u.h[1] = *(const v8h*)(A + m * lda + k0 + 16);
  return u.v;
}
// B 32x16: lanes 0-15 -> N=lane, K chunks {0..7, 16..23}; lanes 16-31 -> K chunks {8..15, 24..31}
// Fed with row-major W[n][k] this computes A @ W^T.
__device__ inline v16h load_b16(const half_t* Bm, int ldb, int nbase, int kbase, int lane) {
  int n  = nbase + (lane & 15);
  int k0 = kbase + ((lane >> 4) << 3);
  const v8h* p = (const v8h*)(Bm + n * ldb + k0);
  V16U u;
  u.h[0] = p[0];
  u.h[1] = p[2];
  return u.v;
}
__device__ inline v8f wmma16(v16h a, v16h b, v8f c) {
  return __builtin_amdgcn_wmma_f32_16x16x32_f16(false, a, false, b, (short)0, c, false, false);
}

// ---- embedding + positional encoding --------------------------------------
__global__ __launch_bounds__(256) void embed_kernel(const int* __restrict__ x,
                                                    const float* __restrict__ emb,
                                                    float* __restrict__ h,
                                                    half_t* __restrict__ h16) {
  int row = blockIdx.x;            // b*S + s
  int d   = threadIdx.x;
  int s   = row & (S_ - 1);
  int tok = x[row];
  int i   = d >> 1;
  float den = __expf(-(float)(2 * i) * (9.210340371976184f / (float)D_)); // ln(10000)
  float ang = (float)s * den;
  float pe  = (d & 1) ? __cosf(ang) : __sinf(ang);
  float v   = emb[tok * D_ + d] + pe;
  h[row * D_ + d]   = v;
  h16[row * D_ + d] = (half_t)v;
}

__global__ void f32_to_f16_kernel(const float* __restrict__ src, half_t* __restrict__ dst, int n) {
  int i = blockIdx.x * blockDim.x + threadIdx.x;
  if (i < n) dst[i] = (half_t)src[i];
}

// ---- generic projection GEMM: out = A[8192x256]f16 @ W[256x256]^T + bias ---
// mode 0: scatter to [B,H,S,HD] (QKV).  mode 1: outF[m*D+n] = acc+bias+resid (O-proj).
__global__ __launch_bounds__(128) void gemm_proj_kernel(const half_t* __restrict__ A,
                                                        const half_t* __restrict__ W,
                                                        const float* __restrict__ bias,
                                                        const float* __restrict__ resid,
                                                        float* __restrict__ outF,
                                                        int mode) {
  int lane = threadIdx.x & 31;
  int w    = threadIdx.x >> 5;
  int mw   = blockIdx.x * 64 + (w & 1) * 32;
  int nw   = blockIdx.y * 64 + (w >> 1) * 32;
  v8f acc[2][2] = {};
  for (int kb = 0; kb < D_; kb += 32) {
    v16h a0 = load_a16(A, D_, mw,      kb, lane);
    v16h a1 = load_a16(A, D_, mw + 16, kb, lane);
    v16h b0 = load_b16(W, D_, nw,      kb, lane);
    v16h b1 = load_b16(W, D_, nw + 16, kb, lane);
    acc[0][0] = wmma16(a0, b0, acc[0][0]);
    acc[0][1] = wmma16(a0, b1, acc[0][1]);
    acc[1][0] = wmma16(a1, b0, acc[1][0]);
    acc[1][1] = wmma16(a1, b1, acc[1][1]);
  }
  int nloc = lane & 15;
  int moff = (lane < 16) ? 0 : 8;
#pragma unroll
  for (int i = 0; i < 2; ++i)
#pragma unroll
    for (int j = 0; j < 2; ++j) {
      int n = nw + j * 16 + nloc;
#pragma unroll
      for (int r = 0; r < 8; ++r) {
        int m = mw + i * 16 + moff + r;
        float val = acc[i][j][r] + bias[n];
        if (mode == 0) {
          int b = m >> 8, s = m & 255, hh = n >> 5, hd = n & 31;
          outF[((b * H_ + hh) * S_ + s) * HD_ + hd] = val;
        } else {
          outF[m * D_ + n] = val + resid[m * D_ + n];
        }
      }
    }
}

// ---- per-head prep: qM=Q@M (f16), qMq, kMk, K f16, V^T f16 -----------------
__global__ __launch_bounds__(256) void qm_kernel(const float* __restrict__ Q,
                                                 const float* __restrict__ K,
                                                 const float* __restrict__ Vv,
                                                 const float* __restrict__ M,
                                                 half_t* __restrict__ qM16,
                                                 half_t* __restrict__ K16,
                                                 half_t* __restrict__ Vt16,
                                                 float* __restrict__ qMq,
                                                 float* __restrict__ kMk) {
  __shared__ float Ms[HD_][HD_];
  int bh = blockIdx.x;
  int s  = threadIdx.x;
  for (int i = threadIdx.x; i < HD_ * HD_; i += 256) ((float*)Ms)[i] = M[i];
  __syncthreads();
  const float* qrow = Q  + (bh * S_ + s) * HD_;
  const float* krow = K  + (bh * S_ + s) * HD_;
  const float* vrow = Vv + (bh * S_ + s) * HD_;
  float q[HD_], k[HD_];
#pragma unroll
  for (int j = 0; j < HD_; ++j) { q[j] = qrow[j]; k[j] = krow[j]; }
  float qq = 0.f, kk = 0.f;
  for (int hd = 0; hd < HD_; ++hd) {
    float vq = 0.f, vk = 0.f;
#pragma unroll
    for (int j = 0; j < HD_; ++j) { vq += q[j] * Ms[j][hd]; vk += k[j] * Ms[j][hd]; }
    qq += q[hd] * vq;
    kk += k[hd] * vk;
    qM16[(bh * S_ + s) * HD_ + hd] = (half_t)vq;
    K16 [(bh * S_ + s) * HD_ + hd] = (half_t)k[hd];
    Vt16[(bh * HD_ + hd) * S_ + s] = (half_t)vrow[hd];
  }
  qMq[bh * S_ + s] = qq;
  kMk[bh * S_ + s] = kk;
}

// ---- scores = -(qMq + kMk - 2*qM@K^T)*sqrt(HD), softmax over keys ----------
__global__ __launch_bounds__(32) void att_score_kernel(const half_t* __restrict__ qM16,
                                                       const half_t* __restrict__ K16,
                                                       const float* __restrict__ qMq,
                                                       const float* __restrict__ kMk,
                                                       float* __restrict__ aw_out,
                                                       half_t* __restrict__ aw16) {
  __shared__ float sc[16][260];
  __shared__ float qMqs[16];
  __shared__ float kMks[256];
  int bh = blockIdx.x;
  int qt = blockIdx.y;
  int lane = threadIdx.x;
  if (lane < 16) qMqs[lane] = qMq[bh * S_ + qt * 16 + lane];
  for (int i = lane; i < 256; i += 32) kMks[i] = kMk[bh * S_ + i];
  __syncthreads();
  const half_t* qA = qM16 + bh * S_ * HD_;
  const half_t* kB = K16  + bh * S_ * HD_;
  v16h a = load_a16(qA, HD_, qt * 16, 0, lane);
  int nloc = lane & 15;
  int moff = (lane < 16) ? 0 : 8;
  for (int nt = 0; nt < 16; ++nt) {
    v8f acc = {};
    v16h b = load_b16(kB, HD_, nt * 16, 0, lane);
    acc = wmma16(a, b, acc);
#pragma unroll
    for (int r = 0; r < 8; ++r) {
      int m = moff + r;
      int n = nt * 16 + nloc;
      sc[m][n] = -(qMqs[m] + kMks[n] - 2.0f * acc[r]) * 5.656854249492381f;
    }
  }
  __syncthreads();
  for (int r = 0; r < 16; ++r) {
    float v[8];
    float mx = -1e30f;
#pragma unroll
    for (int i = 0; i < 8; ++i) { v[i] = sc[r][lane + i * 32]; mx = fmaxf(mx, v[i]); }
    for (int off = 16; off > 0; off >>= 1) mx = fmaxf(mx, __shfl_xor(mx, off, 32));
    float sum = 0.f;
#pragma unroll
    for (int i = 0; i < 8; ++i) { v[i] = __expf(v[i] - mx); sum += v[i]; }
    for (int off = 16; off > 0; off >>= 1) sum += __shfl_xor(sum, off, 32);
    float inv = 1.0f / sum;
    int qrow = qt * 16 + r;
    float*  dst   = aw_out + (bh * S_ + qrow) * S_;
    half_t* dst16 = aw16   + (bh * S_ + qrow) * S_;
#pragma unroll
    for (int i = 0; i < 8; ++i) {
      float p = v[i] * inv;
      dst[lane + i * 32]   = p;
      dst16[lane + i * 32] = (half_t)p;
    }
  }
}

// ---- av = aw @ V (K=256, 8 WMMA k-steps) -----------------------------------
__global__ __launch_bounds__(32) void att_av_kernel(const half_t* __restrict__ aw16,
                                                    const half_t* __restrict__ Vt16,
                                                    float* __restrict__ av,
                                                    half_t* __restrict__ av16) {
  int bh = blockIdx.x;
  int qt = blockIdx.y;
  int lane = threadIdx.x;
  const half_t* A  = aw16 + bh * (S_ * S_);
  const half_t* Bv = Vt16 + bh * (HD_ * S_);
  v8f acc[2] = {};
  for (int ks = 0; ks < S_; ks += 32) {
    v16h a  = load_a16(A, S_, qt * 16, ks, lane);
    v16h b0 = load_b16(Bv, S_, 0,  ks, lane);
    v16h b1 = load_b16(Bv, S_, 16, ks, lane);
    acc[0] = wmma16(a, b0, acc[0]);
    acc[1] = wmma16(a, b1, acc[1]);
  }
  int b = bh >> 3, hh = bh & 7;
  int nloc = lane & 15;
  int moff = (lane < 16) ? 0 : 8;
#pragma unroll
  for (int j = 0; j < 2; ++j)
#pragma unroll
    for (int r = 0; r < 8; ++r) {
      int s  = qt * 16 + moff + r;
      int hd = j * 16 + nloc;
      float val = acc[j][r];
      int idx = (b * S_ + s) * D_ + hh * HD_ + hd;
      av[idx]   = val;
      av16[idx] = (half_t)val;
    }
}

// ---- layernorm -------------------------------------------------------------
__global__ __launch_bounds__(256) void ln_kernel(const float* __restrict__ tmp,
                                                 const float* __restrict__ g,
                                                 const float* __restrict__ bb,
                                                 float* __restrict__ h,
                                                 half_t* __restrict__ h16) {
  __shared__ float s1[256], s2[256];
  int row = blockIdx.x, d = threadIdx.x;
  float v = tmp[row * D_ + d];
  s1[d] = v; s2[d] = v * v;
  __syncthreads();
  for (int off = 128; off > 0; off >>= 1) {
    if (d < off) { s1[d] += s1[d + off]; s2[d] += s2[d + off]; }
    __syncthreads();
  }
  float mu  = s1[0] * (1.0f / D_);
  float var = s2[0] * (1.0f / D_) - mu * mu;
  float inv = rsqrtf(var + 1e-5f);
  float o = (v - mu) * inv * g[d] + bb[d];
  h[row * D_ + d]   = o;
  h16[row * D_ + d] = (half_t)o;
}

// ---- RBF feed-forward ------------------------------------------------------
__global__ __launch_bounds__(256) void rbf_ff_kernel(const float* __restrict__ h,
                                                     const float* __restrict__ centers,
                                                     const float* __restrict__ beta,
                                                     const float* __restrict__ Wfc,
                                                     float* __restrict__ ff) {
  __shared__ float hrow[256];
  __shared__ float red[256];
  __shared__ float rbf[C_];
  int row = blockIdx.x, t = threadIdx.x;
  float v = h[row * D_ + t];
  hrow[t] = v;
  red[t]  = v * v;
  __syncthreads();
  for (int off = 128; off > 0; off >>= 1) {
    if (t < off) red[t] += red[t + off];
    __syncthreads();
  }
  float hh = red[0];
  if (t < C_) {
    const float* c = centers + t * D_;
    float dot = 0.f, cn = 0.f;
    for (int d = 0; d < D_; ++d) { float cv = c[d]; dot += hrow[d] * cv; cn += cv * cv; }
    rbf[t] = __expf(-beta[t] * (hh - 2.0f * dot + cn));
  }
  __syncthreads();
  float acc = 0.f;
  const float* wrow = Wfc + t * C_;
  for (int c = 0; c < C_; ++c) acc += rbf[c] * wrow[c];
  ff[row * D_ + t] = acc;
}

// ---- classifier: logits[b,v] = ff_flat[b,:] . Wout[v,:] + bout[v] ----------
// LDS-tiled so Wout (524 MB) streams exactly once; fp32 for accuracy.
#define KC 256
__global__ __launch_bounds__(256) void logits_kernel(const float* __restrict__ ff,
                                                     const float* __restrict__ Wout,
                                                     const float* __restrict__ bout,
                                                     float* __restrict__ logits) {
  __shared__ float ffS[32][KC + 4];
  __shared__ float woS[16][KC + 4];
  int v0 = blockIdx.x * 16;
  int t  = threadIdx.x;
  int vloc = t >> 4;
  int b0   = t & 15;
  float acc0 = 0.f, acc1 = 0.f;
  for (int k0 = 0; k0 < S_ * D_; k0 += KC) {
    for (int idx = t; idx < 32 * KC; idx += 256) {
      int bb2 = idx >> 8, kk2 = idx & (KC - 1);
      ffS[bb2][kk2] = ff[bb2 * (S_ * D_) + k0 + kk2];
    }
    for (int idx = t; idx < 16 * KC; idx += 256) {
      int vv2 = idx >> 8, kk2 = idx & (KC - 1);
      woS[vv2][kk2] = Wout[(long long)(v0 + vv2) * (S_ * D_) + k0 + kk2];
    }
    __syncthreads();
    for (int kk = 0; kk < KC; ++kk) {
      float w = woS[vloc][kk];
      acc0 += w * ffS[b0][kk];
      acc1 += w * ffS[b0 + 16][kk];
    }
    __syncthreads();
  }
  logits[b0 * V_ + v0 + vloc]        = acc0 + bout[v0 + vloc];
  logits[(b0 + 16) * V_ + v0 + vloc] = acc1 + bout[v0 + vloc];
}

// ---------------------------------------------------------------------------
extern "C" void kernel_launch(void* const* d_in, const int* in_sizes, int n_in,
                              void* d_out, int out_size, void* d_ws, size_t ws_size,
                              hipStream_t stream) {
  (void)in_sizes; (void)n_in; (void)out_size; (void)ws_size;
  const int*   x    = (const int*)  d_in[0];
  const float* emb  = (const float*)d_in[1];
  const float* Wq   = (const float*)d_in[2];
  const float* bq   = (const float*)d_in[3];
  const float* Wk   = (const float*)d_in[4];
  const float* bk   = (const float*)d_in[5];
  const float* Wv   = (const float*)d_in[6];
  const float* bv   = (const float*)d_in[7];
  const float* Mm   = (const float*)d_in[8];
  const float* Wo   = (const float*)d_in[9];
  const float* bo   = (const float*)d_in[10];
  const float* lng  = (const float*)d_in[11];
  const float* lnb  = (const float*)d_in[12];
  const float* cen  = (const float*)d_in[13];
  const float* beta = (const float*)d_in[14];
  const float* Wfc  = (const float*)d_in[15];
  const float* Wout = (const float*)d_in[16];
  const float* bout = (const float*)d_in[17];
  float* out = (float*)d_out;

  char* ws = (char*)d_ws;
  size_t off = 0;
  auto alloc = [&](size_t bytes) -> char* {
    char* p = ws + off;
    off += (bytes + 255) & ~size_t(255);
    return p;
  };
  float*  h    = (float*) alloc((size_t)B_ * S_ * D_ * 4);
  half_t* h16  = (half_t*)alloc((size_t)B_ * S_ * D_ * 2);
  half_t* Wq16 = (half_t*)alloc((size_t)L_ * D_ * D_ * 2);
  half_t* Wk16 = (half_t*)alloc((size_t)L_ * D_ * D_ * 2);
  half_t* Wv16 = (half_t*)alloc((size_t)L_ * D_ * D_ * 2);
  half_t* Wo16 = (half_t*)alloc((size_t)L_ * D_ * D_ * 2);
  float*  Qf   = (float*) alloc((size_t)B_ * S_ * D_ * 4);
  float*  Kf   = (float*) alloc((size_t)B_ * S_ * D_ * 4);
  float*  Vf   = (float*) alloc((size_t)B_ * S_ * D_ * 4);
  half_t* qM16 = (half_t*)alloc((size_t)B_ * H_ * S_ * HD_ * 2);
  half_t* K16  = (half_t*)alloc((size_t)B_ * H_ * S_ * HD_ * 2);
  half_t* Vt16 = (half_t*)alloc((size_t)B_ * H_ * HD_ * S_ * 2);
  float*  qMq  = (float*) alloc((size_t)B_ * H_ * S_ * 4);
  float*  kMk  = (float*) alloc((size_t)B_ * H_ * S_ * 4);
  half_t* aw16 = (half_t*)alloc((size_t)B_ * H_ * S_ * S_ * 2);
  float*  avf  = (float*) alloc((size_t)B_ * S_ * D_ * 4);
  half_t* av16 = (half_t*)alloc((size_t)B_ * S_ * D_ * 2);
  float*  tmp  = (float*) alloc((size_t)B_ * S_ * D_ * 4);
  float*  ffb  = (float*) alloc((size_t)B_ * S_ * D_ * 4);

  embed_kernel<<<dim3(B_ * S_), dim3(D_), 0, stream>>>(x, emb, h, h16);

  int wtot = L_ * D_ * D_;
  f32_to_f16_kernel<<<(wtot + 255) / 256, 256, 0, stream>>>(Wq, Wq16, wtot);
  f32_to_f16_kernel<<<(wtot + 255) / 256, 256, 0, stream>>>(Wk, Wk16, wtot);
  f32_to_f16_kernel<<<(wtot + 255) / 256, 256, 0, stream>>>(Wv, Wv16, wtot);
  f32_to_f16_kernel<<<(wtot + 255) / 256, 256, 0, stream>>>(Wo, Wo16, wtot);

  dim3 ggrid(B_ * S_ / 64, D_ / 64);
  dim3 agrid(B_ * H_, S_ / 16);
  for (int l = 0; l < L_; ++l) {
    const half_t* wq = Wq16 + (size_t)l * D_ * D_;
    const half_t* wk = Wk16 + (size_t)l * D_ * D_;
    const half_t* wv = Wv16 + (size_t)l * D_ * D_;
    const half_t* wo = Wo16 + (size_t)l * D_ * D_;
    gemm_proj_kernel<<<ggrid, 128, 0, stream>>>(h16, wq, bq + l * D_, nullptr, Qf, 0);
    gemm_proj_kernel<<<ggrid, 128, 0, stream>>>(h16, wk, bk + l * D_, nullptr, Kf, 0);
    gemm_proj_kernel<<<ggrid, 128, 0, stream>>>(h16, wv, bv + l * D_, nullptr, Vf, 0);
    qm_kernel<<<B_ * H_, 256, 0, stream>>>(Qf, Kf, Vf, Mm + (size_t)l * HD_ * HD_,
                                           qM16, K16, Vt16, qMq, kMk);
    float* awbase = out + (size_t)B_ * V_ + (size_t)l * B_ * H_ * S_ * S_;
    att_score_kernel<<<agrid, 32, 0, stream>>>(qM16, K16, qMq, kMk, awbase, aw16);
    att_av_kernel<<<agrid, 32, 0, stream>>>(aw16, Vt16, avf, av16);
    gemm_proj_kernel<<<ggrid, 128, 0, stream>>>(av16, wo, bo + l * D_, h, tmp, 1);
    ln_kernel<<<B_ * S_, D_, 0, stream>>>(tmp, lng + l * D_, lnb + l * D_, h, h16);
  }

  rbf_ff_kernel<<<B_ * S_, D_, 0, stream>>>(h, cen, beta, Wfc, ffb);
  logits_kernel<<<V_ / 16, 256, 0, stream>>>(ffb, Wout, bout, out);
}
